// GroupedQueryAttention_76287209112208
// MI455X (gfx1250) — matrix operand order repaired and checked by reading, not compiled, not run on hardware
//
#include <hip/hip_runtime.h>
#include <math.h>

// ---------------------------------------------------------------------------
// GQA forward for MI455X (gfx1250, wave32, WMMA).
// Shapes: B=2, S=2048, D=1024, HQ=16, HKV=4, HD=64, DKV=256, G=4.
// Outputs (concatenated in d_out): out [B,S,D] fp32, attn [B,HQ,S,S] fp32.
// bf16 storage for all GEMM operands (converted once); WMMA bf16 matmuls;
// fused scores+softmax+AV kernel keeps the 16x2048 prob strip in LDS
// (320KB/WGP) so attn is written to HBM exactly once, vectorized b128.
// ---------------------------------------------------------------------------

#define BB   2
#define SS   2048
#define DD   1024
#define HQ   16
#define HKV  4
#define HD   64
#define DKV  256
#define GG   (HQ / HKV)
#define SCALE 0.125f            // 1/sqrt(64)
#define MROWS (BB * SS)         // 4096
#define PROBS_LD (SS + 4)       // +4 f32 pad: row base stays 16B-aligned,
                                // AV row gathers spread across banks

typedef __attribute__((ext_vector_type(16))) __bf16 v16bf;
typedef __attribute__((ext_vector_type(8)))  float  v8f;
typedef int b128v __attribute__((__vector_size__(16)));   // matches async builtin

// --- async copy global->LDS (CDNA5 GLOBAL_LOAD_ASYNC_TO_LDS_B128) ----------
#if defined(__has_builtin)
#if __has_builtin(__builtin_amdgcn_global_load_async_to_lds_b128)
#define HAVE_ASYNC_LDS 1
#endif
#endif

__device__ __forceinline__ void async_copy_16B(const void* g, void* l) {
#ifdef HAVE_ASYNC_LDS
  __builtin_amdgcn_global_load_async_to_lds_b128(
      (__attribute__((address_space(1))) b128v*)g,
      (__attribute__((address_space(3))) b128v*)l, 0, 0);
#else
  *(float4*)l = *(const float4*)g;
#endif
}

__device__ __forceinline__ void async_join() {
#ifdef HAVE_ASYNC_LDS
#if __has_builtin(__builtin_amdgcn_s_wait_asynccnt)
  __builtin_amdgcn_s_wait_asynccnt(0);
#else
  asm volatile("s_wait_asynccnt 0x0" ::: "memory");
#endif
#endif
}

// --- WMMA fragment helpers (A/B use contiguous k = 16*(lane/16) + s) -------

// A fragment from row-major bf16: lane<16: M=lane, K=k0..+15;
// lane>=16: M=lane-16, K=k0+16..+31.  Two contiguous 16B loads.
__device__ __forceinline__ v16bf load_a_frag(const __bf16* __restrict__ A,
                                             int lda, int row0, int k0, int lane) {
  int r  = row0 + (lane & 15);
  int kk = k0 + ((lane >> 4) ? 16 : 0);
  const __bf16* p = A + (size_t)r * lda + kk;
  v16bf a;
#pragma unroll
  for (int i = 0; i < 8; ++i) a[i] = p[i];
#pragma unroll
  for (int i = 0; i < 8; ++i) a[8 + i] = p[8 + i];
  return a;
}

// A fragment from an f32 source with arbitrary row stride (LDS probs strip).
__device__ __forceinline__ v16bf load_a_frag_f32(const float* __restrict__ A,
                                                 int lda, int k0, int lane) {
  int r  = lane & 15;
  int kk = k0 + ((lane >> 4) ? 16 : 0);
  const float* p = A + (size_t)r * lda + kk;
  v16bf a;
#pragma unroll
  for (int i = 0; i < 8; ++i) a[i] = (__bf16)p[i];
#pragma unroll
  for (int i = 0; i < 8; ++i) a[8 + i] = (__bf16)p[8 + i];
  return a;
}

// B fragment where the source is row-major [N, K] bf16 (A @ src^T).
// Each lane loads 16 contiguous bf16 (32B).
__device__ __forceinline__ v16bf load_bt_frag(const __bf16* __restrict__ Bt,
                                              int ldb, int n0, int k0, int lane) {
  int n  = n0 + (lane & 15);
  int kk = k0 + ((lane >> 4) ? 16 : 0);
  const __bf16* p = Bt + (size_t)n * ldb + kk;
  v16bf b;
#pragma unroll
  for (int i = 0; i < 16; ++i) b[i] = p[i];
  return b;
}

// B fragment gathered from an LDS tile Bs[32][64] bf16, columns n0..n0+15.
__device__ __forceinline__ v16bf lds_b_frag(const __bf16* Bs, int n0, int lane) {
  int n  = n0 + (lane & 15);
  int kk = (lane >> 4) ? 16 : 0;
  const __bf16* p = Bs + (size_t)kk * 64 + n;
  v16bf b;
#pragma unroll
  for (int i = 0; i < 16; ++i) b[i] = p[i * 64];
  return b;
}

// C store helpers: f32 16x16 layout — VGPR r: M=r (lanes 0-15) / M=r+8 (16-31).
__device__ __forceinline__ void store_elem(float* C, size_t i, float v)  { C[i] = v; }
__device__ __forceinline__ void store_elem(__bf16* C, size_t i, float v) { C[i] = (__bf16)v; }

template <typename OT>
__device__ __forceinline__ void store_c(OT* __restrict__ C, int ldc,
                                        int row0, int col0, v8f c, int lane,
                                        float alpha, const float* __restrict__ bias) {
  int col   = col0 + (lane & 15);
  int rbase = row0 + ((lane >> 4) ? 8 : 0);
  float bv = bias ? bias[col] : 0.0f;
#pragma unroll
  for (int r = 0; r < 8; ++r)
    store_elem(C, (size_t)(rbase + r) * ldc + col, c[r] * alpha + bv);
}

#define WMMA_BF16(a, b, acc) \
  __builtin_amdgcn_wmma_f32_16x16x32_bf16(false, (a), false, (b), (short)0, (acc), false, false)

// --- shared block GEMM: 64x64 C tile per 128-thread block ------------------
template <typename AT, typename OT>
__device__ __forceinline__ void gemm_block_64x64(
    const AT* __restrict__ A, int lda,
    const __bf16* __restrict__ Bm, int ldb,
    OT* __restrict__ C, int ldc, int K,
    const float* __restrict__ bias, float alpha) {
  __shared__ __bf16 Bs[32 * 64];                // 4 KB
  const int tid  = threadIdx.x;
  const int lane = tid & 31;
  const int wave = tid >> 5;
  const int row0 = wave * 16;
  const int srow = tid >> 2;                    // staging: 32 rows
  const int scol = (tid & 3) * 16;              // 16 bf16 = 32B per thread

  v8f acc0 = {}, acc1 = {}, acc2 = {}, acc3 = {};
  for (int k0 = 0; k0 < K; k0 += 32) {
    const __bf16* gsrc = Bm + (size_t)(k0 + srow) * ldb + scol;
    __bf16* ldst = &Bs[srow * 64 + scol];
    async_copy_16B(gsrc,     ldst);
    async_copy_16B(gsrc + 8, ldst + 8);
    async_join();
    __syncthreads();

    v16bf a  = load_a_frag(A, lda, row0, k0, lane);
    v16bf b0 = lds_b_frag(Bs, 0,  lane);
    v16bf b1 = lds_b_frag(Bs, 16, lane);
    v16bf b2 = lds_b_frag(Bs, 32, lane);
    v16bf b3 = lds_b_frag(Bs, 48, lane);
    acc0 = WMMA_BF16(a, b0, acc0);
    acc1 = WMMA_BF16(a, b1, acc1);
    acc2 = WMMA_BF16(a, b2, acc2);
    acc3 = WMMA_BF16(a, b3, acc3);
    __syncthreads();
  }
  store_c(C, ldc, row0, 0,  acc0, lane, alpha, bias);
  store_c(C, ldc, row0, 16, acc1, lane, alpha, bias);
  store_c(C, ldc, row0, 32, acc2, lane, alpha, bias);
  store_c(C, ldc, row0, 48, acc3, lane, alpha, bias);
}

// --- convert f32 -> bf16 ---------------------------------------------------
__global__ __launch_bounds__(256) void cvt_f32_bf16(
    const float* __restrict__ src, __bf16* __restrict__ dst, int n) {
  int i = (blockIdx.x * 256 + threadIdx.x) * 8;
  if (i >= n) return;
#pragma unroll
  for (int j = 0; j < 8; ++j) dst[i + j] = (__bf16)src[i + j];
}

// --- GEMM + bias: C[M,N] = A[M,K] @ W[K,N] + bias --------------------------
template <typename AT, typename OT>
__global__ __launch_bounds__(128) void gemm_bias_64x64(
    const AT* __restrict__ A, const __bf16* __restrict__ W,
    const float* __restrict__ bias, OT* __restrict__ C, int N, int K) {
  int nb = N >> 6;
  int br = blockIdx.x / nb;
  int cb = blockIdx.x % nb;
  gemm_block_64x64(A + (size_t)br * 64 * K, K,
                   W + cb * 64, N,
                   C + (size_t)br * 64 * N + cb * 64, N,
                   K, bias + cb * 64, 1.0f);
}

// --- fused attention: scores -> softmax -> attn@V per (b,h,16-row strip) ---
// Block = 128 threads (4 waves). LDS: probs[16][PROBS_LD] f32 (~128KB) +
// 16KB async-staged V tile; ~144KB total -> 2 blocks per 320KB WGP.
// attn is written to HBM exactly once, fully vectorized (b128).
__global__ __launch_bounds__(128) void gqa_fused(
    const __bf16* __restrict__ qp, const __bf16* __restrict__ kp,
    const __bf16* __restrict__ vp, float* __restrict__ attn,
    __bf16* __restrict__ ctx) {
  __shared__ float  probs[16 * PROBS_LD];       // 16 x 2048 (+pad) f32
  __shared__ __bf16 Bs[128 * 64];               // V tile, 16KB
  const int tid  = threadIdx.x;
  const int lane = tid & 31;
  const int wave = tid >> 5;
  const int ti = blockIdx.x & 127;              // 16-row strip index
  const int bh = blockIdx.x >> 7;               // 0..B*HQ-1
  const int b = bh / HQ, h = bh % HQ, kvh = h / GG;

  // ---- phase 1: scores -> LDS.  Wave w covers j in [w*512, w*512+512). ----
  const __bf16* Aq = qp + (size_t)b * SS * DD  + (size_t)h   * HD;  // lda=D
  const __bf16* Bk = kp + (size_t)b * SS * DKV + (size_t)kvh * HD;  // [S,HD]
  v16bf qa0 = load_a_frag(Aq, DD, ti * 16, 0,  lane);
  v16bf qa1 = load_a_frag(Aq, DD, ti * 16, 32, lane);
  const int rbase = (lane >> 4) ? 8 : 0;
#pragma unroll 4
  for (int jt = 0; jt < 32; ++jt) {
    int j0 = wave * 512 + jt * 16;
    v16bf bt0 = load_bt_frag(Bk, DKV, j0, 0,  lane);
    v16bf bt1 = load_bt_frag(Bk, DKV, j0, 32, lane);
    v8f acc = {};
    acc = WMMA_BF16(qa0, bt0, acc);
    acc = WMMA_BF16(qa1, bt1, acc);
    int col = j0 + (lane & 15);
#pragma unroll
    for (int r = 0; r < 8; ++r)
      probs[(rbase + r) * PROBS_LD + col] = acc[r] * SCALE;
  }
  __syncthreads();

  // ---- phase 2: row softmax in LDS; stream probs to attn (b128). ----------
  for (int rr = 0; rr < 4; ++rr) {
    int row = wave * 4 + rr;
    float4* pr4 = (float4*)&probs[row * PROBS_LD];     // 16B-aligned rows
    float m = -INFINITY;
    for (int c = lane; c < SS / 4; c += 32) {
      float4 v = pr4[c];
      m = fmaxf(m, fmaxf(fmaxf(v.x, v.y), fmaxf(v.z, v.w)));
    }
#pragma unroll
    for (int off = 16; off; off >>= 1) m = fmaxf(m, __shfl_xor(m, off, 32));
    float sum = 0.0f;
    for (int c = lane; c < SS / 4; c += 32) {
      float4 v = pr4[c];
      v.x = __expf(v.x - m); v.y = __expf(v.y - m);
      v.z = __expf(v.z - m); v.w = __expf(v.w - m);
      sum += (v.x + v.y) + (v.z + v.w);
      pr4[c] = v;
    }
#pragma unroll
    for (int off = 16; off; off >>= 1) sum += __shfl_xor(sum, off, 32);
    float inv = 1.0f / sum;
    float4* g4 = (float4*)(attn + ((size_t)bh * SS + (size_t)(ti * 16 + row)) * SS);
    for (int c = lane; c < SS / 4; c += 32) {
      float4 v = pr4[c];
      v.x *= inv; v.y *= inv; v.z *= inv; v.w *= inv;
      pr4[c] = v;
      g4[c] = v;                                // the single HBM attn write
    }
  }
  __syncthreads();

  // ---- phase 3: ctx[16 x 64] = probs[16 x S] @ vh[S x 64]. ----------------
  // Wave w owns output columns w*16..w*16+15. 128x64 V tile async-staged
  // per stage (one row per thread, 8 x b128); 4 WMMAs per barrier pair.
  const __bf16* Bv = vp + (size_t)b * SS * DKV + (size_t)kvh * HD;  // [S,HD]
  v8f acc = {};
  for (int k0 = 0; k0 < SS; k0 += 128) {
    const __bf16* grow = Bv + (size_t)(k0 + tid) * DKV;
    __bf16* lrow = &Bs[tid * 64];
#pragma unroll
    for (int j = 0; j < 8; ++j)
      async_copy_16B(grow + j * 8, lrow + j * 8);
    async_join();
    __syncthreads();
#pragma unroll
    for (int kk = 0; kk < 128; kk += 32) {
      v16bf a  = load_a_frag_f32(probs, PROBS_LD, k0 + kk, lane);
      v16bf bf = lds_b_frag(Bs + kk * 64, wave * 16, lane);
      acc = WMMA_BF16(a, bf, acc);
    }
    __syncthreads();
  }
  __bf16* Cp = ctx + (size_t)b * SS * DD + (size_t)h * HD;
  store_c(Cp, DD, ti * 16, wave * 16, acc, lane, 1.0f, nullptr);
}

// ---------------------------------------------------------------------------
extern "C" void kernel_launch(void* const* d_in, const int* in_sizes, int n_in,
                              void* d_out, int out_size, void* d_ws, size_t ws_size,
                              hipStream_t stream) {
  const float* q  = (const float*)d_in[0];
  const float* Wq = (const float*)d_in[1];
  const float* bq = (const float*)d_in[2];
  const float* Wk = (const float*)d_in[3];
  const float* bk = (const float*)d_in[4];
  const float* Wv = (const float*)d_in[5];
  const float* bv = (const float*)d_in[6];
  const float* Wo = (const float*)d_in[7];
  const float* bo = (const float*)d_in[8];

  float* out  = (float*)d_out;                       // [B,S,D]
  float* attn = out + (size_t)BB * SS * DD;          // [B,HQ,S,S]

  // bf16 workspace slabs (elements)
  const size_t nQ  = (size_t)BB * SS * DD;           // 4.19M
  const size_t nWq = (size_t)DD * DD;                // 1.05M
  const size_t nWk = (size_t)DD * DKV;               // 0.26M
  const size_t nKP = (size_t)BB * SS * DKV;          // 1.05M
  __bf16* qb  = (__bf16*)d_ws;
  __bf16* Wqb = qb  + nQ;
  __bf16* Wkb = Wqb + nWq;
  __bf16* Wvb = Wkb + nWk;
  __bf16* Wob = Wvb + nWk;
  __bf16* qp  = Wob + nWq;                           // [B,S,D] bf16
  __bf16* kp  = qp  + nQ;                            // [B,S,DKV] bf16
  __bf16* vp  = kp  + nKP;                           // [B,S,DKV] bf16
  __bf16* ctx = vp  + nKP;                           // [B,S,D] bf16

  dim3 b128t(128), b256t(256);
  // one-time f32 -> bf16 conversions (bandwidth-trivial)
  cvt_f32_bf16<<<nQ  / 2048, b256t, 0, stream>>>(q,  qb,  (int)nQ);
  cvt_f32_bf16<<<nWq / 2048, b256t, 0, stream>>>(Wq, Wqb, (int)nWq);
  cvt_f32_bf16<<<nWk / 2048, b256t, 0, stream>>>(Wk, Wkb, (int)nWk);
  cvt_f32_bf16<<<nWk / 2048, b256t, 0, stream>>>(Wv, Wvb, (int)nWk);
  cvt_f32_bf16<<<nWq / 2048, b256t, 0, stream>>>(Wo, Wob, (int)nWq);

  // Q/K/V projections: bf16 x bf16 -> bf16 (+f32 bias)
  gemm_bias_64x64<<<(MROWS / 64) * (DD  / 64), b128t, 0, stream>>>(qb, Wqb, bq, qp, DD,  DD);
  gemm_bias_64x64<<<(MROWS / 64) * (DKV / 64), b128t, 0, stream>>>(qb, Wkb, bk, kp, DKV, DD);
  gemm_bias_64x64<<<(MROWS / 64) * (DKV / 64), b128t, 0, stream>>>(qb, Wvb, bv, vp, DKV, DD);
  // fused scores + softmax + attn@V  (attn written to HBM exactly once)
  gqa_fused<<<BB * HQ * (SS / 16), b128t, 0, stream>>>(qp, kp, vp, attn, ctx);
  // output projection: bf16 x bf16 -> f32 out
  gemm_bias_64x64<<<(MROWS / 64) * (DD / 64), b128t, 0, stream>>>(ctx, Wob, bo, out, DD, DD);
}
